// MultilayerGRU_90580860272912
// MI455X (gfx1250) — matrix operand order repaired and checked by reading, not compiled, not run on hardware
//
#include <hip/hip_runtime.h>
#include <hip/hip_bf16.h>

#define B_ 32
#define S_ 2048
#define I_ 128
#define H_ 512
#define O_ 128
#define G3_ (3*H_)
#define NBLK 16
#define THR 128

typedef _Float16 v16h __attribute__((ext_vector_type(16)));
typedef _Float16 v8h  __attribute__((ext_vector_type(8)));
typedef float    v8f  __attribute__((ext_vector_type(8)));

__device__ __forceinline__ v8f wmma_f16(v16h a, v16h b, v8f c) {
  // D = A(16x32 f16) * B(32x16 f16) + C(16x16 f32)
  return __builtin_amdgcn_wmma_f32_16x16x32_f16(
      /*neg_a=*/false, a, /*neg_b=*/false, b,
      /*c_mod=*/(short)0, c, /*reuse_a=*/false, /*reuse_b=*/false);
}

// A fragment: 16x32 f16 tile from row-major [rows x ld], rows m0..m0+15, cols k..k+31.
// ISA layout: lane L holds M=L%16; elems 0..7 = K = (L/16)*8 + 0..7; elems 8..15 = +16.
__device__ __forceinline__ v16h loadA(const _Float16* base, int ld,
                                      int m0, int k, int lane) {
  int row = m0 + (lane & 15);
  int kh  = k + ((lane >> 4) << 3);
  const _Float16* p = base + (size_t)row * ld + kh;
  v8h lo = *(const v8h*)(p);
  v8h hi = *(const v8h*)(p + 16);
  return __builtin_shufflevector(lo, hi, 0,1,2,3,4,5,6,7,8,9,10,11,12,13,14,15);
}

// B fragment: 32x16 f16 tile = transpose-view of row-major weight W[n][k].
// ISA layout: lane L holds N=L%16; elems 0..7 = K = (L/16)*8 + 0..7; elems 8..15 = +16.
// => per lane: two 8-f16 runs from weight row (n0 + L%16).
__device__ __forceinline__ v16h loadB(const _Float16* w, int ld,
                                      int n0, int k, int lane) {
  int row = n0 + (lane & 15);
  int kk  = k + ((lane >> 4) << 3);
  const _Float16* p = w + (size_t)row * ld + kk;
  v8h lo = *(const v8h*)(p);
  v8h hi = *(const v8h*)(p + 16);
  return __builtin_shufflevector(lo, hi, 0,1,2,3,4,5,6,7,8,9,10,11,12,13,14,15);
}

__device__ __forceinline__ float sigmoidf_(float x) { return 1.0f / (1.0f + __expf(-x)); }

// Sense-reversing grid-wide barrier (all NBLK blocks resident/persistent).
__device__ __forceinline__ void gridsync(unsigned* bar) {
  __threadfence();
  __syncthreads();
  if (threadIdx.x == 0) {
    volatile unsigned* cnt = bar;
    volatile unsigned* gen = bar + 1;
    unsigned g = *gen;
    if (atomicAdd(bar, 1u) == NBLK - 1) {
      *cnt = 0u;
      __threadfence();
      *gen = g + 1u;
    } else {
      while (*gen == g) { __builtin_amdgcn_s_sleep(1); }
    }
  }
  __syncthreads();
  __threadfence();
}

__global__ void cvt_f32_f16(const float* __restrict__ s, _Float16* __restrict__ d, int n) {
  int i = blockIdx.x * blockDim.x + threadIdx.x;
  if (i < n) d[i] = (_Float16)s[i];
}

__global__ void bar_init(unsigned* bar) {
  if (threadIdx.x == 0 && blockIdx.x == 0) { bar[0] = 0u; bar[1] = 0u; }
}

// Persistent 2-layer GRU scan. 64 waves: wave gw -> row-tile m0=(gw&1)*16,
// column-group cg=gw>>1 owning hidden columns [cg*16, cg*16+16) of z,r,g gates.
// NOTE: weight/h pointers deliberately NOT __restrict__ so weight-fragment loads
// cannot be hoisted out of the sequence loop (which caused scratch spills).
__global__ __launch_bounds__(THR, 1) void gru_scan(
    const _Float16* x16,                                   // [B][S][I]
    const _Float16* wx0, const _Float16* wh0,              // [3H][I], [3H][H]
    const _Float16* wx1, const _Float16* wh1,              // [3H][H]
    const _Float16* wy,                                    // [O][H]
    const float* bx0, const float* bx1, const float* by,
    _Float16* h0buf, _Float16* h1buf,                      // [2][B][H] ping-pong
    float* __restrict__ out, unsigned* bar)
{
  const int lane = threadIdx.x & 31;
  const int gw   = blockIdx.x * (THR / 32) + (threadIdx.x >> 5);
  const int m0   = (gw & 1) * 16;      // batch-row tile base
  const int cg   = gw >> 1;            // 0..31 hidden-column group
  const int half = lane >> 4;
  const int n    = lane & 15;
  const int c    = cg * 16 + n;        // hidden column 0..511
  const int zrow = cg * 16, rrow = H_ + cg * 16, grow = 2 * H_ + cg * 16;

  // zero both ping-pong h buffers
  for (int i = blockIdx.x * THR + threadIdx.x; i < 2 * B_ * H_; i += NBLK * THR) {
    h0buf[i] = (_Float16)0.0f;
    h1buf[i] = (_Float16)0.0f;
  }
  gridsync(bar);

  const float bz0 = bx0[c], br0 = bx0[H_ + c], bg0 = bx0[2 * H_ + c];
  const float bz1 = bx1[c], br1 = bx1[H_ + c], bg1 = bx1[2 * H_ + c];
  const float byc = (cg < O_ / 16) ? by[cg * 16 + n] : 0.0f;

  for (int s = 0; s < S_; ++s) {
    const int pin = s & 1, pout = pin ^ 1;
    const _Float16* h0i = h0buf + pin  * (B_ * H_);
    _Float16*       h0o = h0buf + pout * (B_ * H_);
    const _Float16* h1i = h1buf + pin  * (B_ * H_);
    _Float16*       h1o = h1buf + pout * (B_ * H_);

    // ---------------- layer 0: gates = [x_t,h0] . [wx0;wh0]^T + bx0 ----------------
    v8f az, ar, axg, ahg;
    #pragma unroll
    for (int v = 0; v < 8; ++v) { az[v] = bz0; ar[v] = br0; axg[v] = bg0; ahg[v] = 0.0f; }
    const _Float16* xs = x16 + (size_t)s * I_;   // row b at stride S_*I_
    #pragma unroll 1
    for (int k = 0; k < I_; k += 32) {
      v16h a = loadA(xs, S_ * I_, m0, k, lane);
      az  = wmma_f16(a, loadB(wx0, I_, zrow, k, lane), az);
      ar  = wmma_f16(a, loadB(wx0, I_, rrow, k, lane), ar);
      axg = wmma_f16(a, loadB(wx0, I_, grow, k, lane), axg);
    }
    #pragma unroll 1
    for (int k = 0; k < H_; k += 32) {
      v16h a = loadA(h0i, H_, m0, k, lane);
      az  = wmma_f16(a, loadB(wh0, H_, zrow, k, lane), az);
      ar  = wmma_f16(a, loadB(wh0, H_, rrow, k, lane), ar);
      ahg = wmma_f16(a, loadB(wh0, H_, grow, k, lane), ahg);
    }
    #pragma unroll
    for (int v = 0; v < 8; ++v) {
      int row = m0 + half * 8 + v;
      float z = sigmoidf_(az[v]);
      float r = sigmoidf_(ar[v]);
      float g = tanhf(axg[v] + r * ahg[v]);
      float hprev = (float)h0i[(size_t)row * H_ + c];
      h0o[(size_t)row * H_ + c] = (_Float16)(g + z * (hprev - g));
    }
    gridsync(bar);   // h0(t) visible device-wide

    // ---------------- layer 1: gates = [h0,h1] . [wx1;wh1]^T + bx1 ----------------
    #pragma unroll
    for (int v = 0; v < 8; ++v) { az[v] = bz1; ar[v] = br1; axg[v] = bg1; ahg[v] = 0.0f; }
    #pragma unroll 1
    for (int k = 0; k < H_; k += 32) {
      v16h a0 = loadA(h0o, H_, m0, k, lane);
      v16h a1 = loadA(h1i, H_, m0, k, lane);
      az  = wmma_f16(a0, loadB(wx1, H_, zrow, k, lane), az);
      ar  = wmma_f16(a0, loadB(wx1, H_, rrow, k, lane), ar);
      axg = wmma_f16(a0, loadB(wx1, H_, grow, k, lane), axg);
      az  = wmma_f16(a1, loadB(wh1, H_, zrow, k, lane), az);
      ar  = wmma_f16(a1, loadB(wh1, H_, rrow, k, lane), ar);
      ahg = wmma_f16(a1, loadB(wh1, H_, grow, k, lane), ahg);
    }
    #pragma unroll
    for (int v = 0; v < 8; ++v) {
      int row = m0 + half * 8 + v;
      float z = sigmoidf_(az[v]);
      float r = sigmoidf_(ar[v]);
      float g = tanhf(axg[v] + r * ahg[v]);
      float hprev = (float)h1i[(size_t)row * H_ + c];
      h1o[(size_t)row * H_ + c] = (_Float16)(g + z * (hprev - g));
    }
    gridsync(bar);   // h1(t) visible device-wide

    // ---------------- y = h1 . wy^T + by (waves with cg < 8) ----------------
    if (cg < O_ / 16) {               // wave-uniform branch: EXEC stays all-1s for WMMA
      v8f acc;
      #pragma unroll
      for (int v = 0; v < 8; ++v) acc[v] = byc;
      #pragma unroll 1
      for (int k = 0; k < H_; k += 32) {
        v16h a = loadA(h1o, H_, m0, k, lane);
        acc = wmma_f16(a, loadB(wy, H_, cg * 16, k, lane), acc);
      }
      #pragma unroll
      for (int v = 0; v < 8; ++v) {
        int row = m0 + half * 8 + v;   // batch index
        out[(size_t)row * (S_ * O_) + (size_t)s * O_ + cg * 16 + n] = acc[v];
      }
    }
    // no barrier needed here: next step's h0[pin] writes were last read at this
    // step's first barrier; h1o is not overwritten until step s+2's layer-1
    // phase, which is behind the next gridsync.
  }

  // final hidden state -> out + B*S*O, layout (B, 2, H)
  {
    const int fo = ((S_ - 1) & 1) ^ 1;
    const _Float16* h0f = h0buf + fo * (B_ * H_);
    const _Float16* h1f = h1buf + fo * (B_ * H_);
    float* hid = out + (size_t)B_ * S_ * O_;
    for (int i = blockIdx.x * THR + threadIdx.x; i < B_ * 2 * H_; i += NBLK * THR) {
      int b = i / (2 * H_);
      int rem = i % (2 * H_);
      int l = rem / H_;
      int j = rem % H_;
      hid[i] = (float)((l == 0 ? h0f : h1f)[b * H_ + j]);
    }
  }
}

extern "C" void kernel_launch(void* const* d_in, const int* in_sizes, int n_in,
                              void* d_out, int out_size, void* d_ws, size_t ws_size,
                              hipStream_t stream) {
  const float* x   = (const float*)d_in[0];
  const float* wx0 = (const float*)d_in[1];
  const float* bx0 = (const float*)d_in[2];
  const float* wh0 = (const float*)d_in[3];
  const float* wx1 = (const float*)d_in[4];
  const float* bx1 = (const float*)d_in[5];
  const float* wh1 = (const float*)d_in[6];
  const float* wy  = (const float*)d_in[7];
  const float* by  = (const float*)d_in[8];
  float* out = (float*)d_out;

  char* ws = (char*)d_ws;
  size_t off = 0;
  auto alloc = [&](size_t bytes) -> char* {
    char* p = ws + off;
    off += (bytes + 255) & ~(size_t)255;
    return p;
  };
  _Float16* x16   = (_Float16*)alloc((size_t)B_ * S_ * I_ * 2);
  _Float16* wx0h  = (_Float16*)alloc((size_t)G3_ * I_ * 2);
  _Float16* wh0h  = (_Float16*)alloc((size_t)G3_ * H_ * 2);
  _Float16* wx1h  = (_Float16*)alloc((size_t)G3_ * H_ * 2);
  _Float16* wh1h  = (_Float16*)alloc((size_t)G3_ * H_ * 2);
  _Float16* wyh   = (_Float16*)alloc((size_t)O_ * H_ * 2);
  _Float16* h0buf = (_Float16*)alloc((size_t)2 * B_ * H_ * 2);
  _Float16* h1buf = (_Float16*)alloc((size_t)2 * B_ * H_ * 2);
  unsigned* bar   = (unsigned*)alloc(256);

  auto launch_cvt = [&](const float* s, _Float16* d, int n) {
    cvt_f32_f16<<<(n + 255) / 256, 256, 0, stream>>>(s, d, n);
  };
  launch_cvt(x,   x16,  B_ * S_ * I_);
  launch_cvt(wx0, wx0h, G3_ * I_);
  launch_cvt(wh0, wh0h, G3_ * H_);
  launch_cvt(wx1, wx1h, G3_ * H_);
  launch_cvt(wh1, wh1h, G3_ * H_);
  launch_cvt(wy,  wyh,  O_ * H_);
  bar_init<<<1, 32, 0, stream>>>(bar);

  gru_scan<<<NBLK, THR, 0, stream>>>(x16, wx0h, wh0h, wx1h, wh1h, wyh,
                                     bx0, bx1, by, h0buf, h1buf, out, bar);
  (void)in_sizes; (void)n_in; (void)out_size; (void)ws_size;
}
